// MultiHeadAttn_14611478741643
// MI455X (gfx1250) — compile-verified, matrix operand order checked by reading, not run on hardware
//
#include <hip/hip_runtime.h>
#include <hip/hip_bf16.h>

// Problem constants (from reference)
#define D_MODEL  1024
#define N_HEAD   16
#define HEAD_DIM 64
#define SEQ      2048
#define BATCH    4

typedef __bf16 bf16_t;
typedef __bf16 v16bf __attribute__((ext_vector_type(16)));
typedef __bf16 v8bf  __attribute__((ext_vector_type(8)));
typedef float  v8f   __attribute__((ext_vector_type(8)));
typedef unsigned int u32x4 __attribute__((ext_vector_type(4)));
typedef int          i32x4 __attribute__((ext_vector_type(4)));
typedef int          i32x8 __attribute__((ext_vector_type(8)));

// fp32 -> bf16 round-to-nearest-even
static __device__ inline bf16_t f2bf(float f) {
  unsigned u = __builtin_bit_cast(unsigned, f);
  u += 0x7FFFu + ((u >> 16) & 1u);
  unsigned short h = (unsigned short)(u >> 16);
  return __builtin_bit_cast(bf16_t, h);
}

static __device__ inline v8f zero8() {
  v8f z;
#pragma unroll
  for (int i = 0; i < 8; ++i) z[i] = 0.0f;
  return z;
}

// Load a 16x32 bf16 A-fragment (or the matching B-fragment, which for our
// row-major "other matrix" layouts uses identical addressing) from a
// row-major bf16 buffer with leading dimension `ld`.
static __device__ inline v16bf load_frag(const bf16_t* base, int ld, int lane) {
  const int row = lane & 15;
  const int kh  = (lane >> 4) << 3;  // 0 or 8
  const bf16_t* p = base + (size_t)row * ld + kh;
  v8bf lo = *(const v8bf*)(p);
  v8bf hi = *(const v8bf*)(p + 16);
  v16bf out;
#pragma unroll
  for (int i = 0; i < 8; ++i) { out[i] = lo[i]; out[i + 8] = hi[i]; }
  return out;
}

static __device__ inline v8f wmma_bf16(v16bf a, v16bf b, v8f c) {
  return __builtin_amdgcn_wmma_f32_16x16x32_bf16(
      /*neg_a=*/false, a, /*neg_b=*/false, b,
      /*c_mod=*/(short)0, c, /*reuse_a=*/false, /*reuse_b=*/false);
}

#if defined(__HIP_DEVICE_COMPILE__) && \
    __has_builtin(__builtin_amdgcn_tensor_load_to_lds) && \
    __has_builtin(__builtin_amdgcn_s_wait_tensorcnt)
#define USE_TDM 1
#endif

#ifdef USE_TDM
// Issue one TDM DMA of a 2D bf16 tile [128 rows x 64 cols] from global memory
// (row stride 1024 elements) into LDS.  Caller must be wave-uniform.
static __device__ inline void tdm_load_tile(const bf16_t* gptr, unsigned lds_addr,
                                            int tensor_rows) {
  const unsigned long long ga = (unsigned long long)(size_t)gptr;
  u32x4 g0;
  g0[0] = 1u;                                   // count=1, user descriptor
  g0[1] = lds_addr;                             // LDS byte address
  g0[2] = (unsigned)ga;                         // global_addr[31:0]
  g0[3] = (unsigned)((ga >> 32) & 0x1FFFFFFu)   // global_addr[56:32]
        | (2u << 30);                           // type=2 ("image")
  i32x8 g1;
  g1[0] = 1 << 16;                              // data_size = 2 bytes
  g1[1] = (D_MODEL & 0xFFFF) << 16;             // tensor_dim0 = 1024
  g1[2] = (tensor_rows & 0xFFFF) << 16;         // tensor_dim1 low bits
  g1[3] = (64 << 16) | ((tensor_rows >> 16) & 0xFFFF);  // tile_dim0 = 64
  g1[4] = 128;                                  // tile_dim1 = 128
  g1[5] = D_MODEL;                              // tensor_dim0_stride = 1024
  g1[6] = 0;
  g1[7] = 0;
  i32x4 g2 = {0, 0, 0, 0}, g3 = {0, 0, 0, 0};   // 2D tile: groups 2/3 unused
#if __clang_major__ >= 23
  i32x8 g4 = {0, 0, 0, 0, 0, 0, 0, 0};
  __builtin_amdgcn_tensor_load_to_lds(g0, g1, g2, g3, g4, 0);
#else
  __builtin_amdgcn_tensor_load_to_lds(g0, g1, g2, g3, 0);
#endif
}
#endif

// ---------------------------------------------------------------------------
// Kernel 0: fp32 -> bf16 bulk convert (one pass; removes all conversion work
// from the GEMM hot loop and enables TDM staging of bf16 tiles).
// ---------------------------------------------------------------------------
__global__ __launch_bounds__(256) void cvt_kernel(
    const float* __restrict__ src, bf16_t* __restrict__ dst)
{
  const size_t i = ((size_t)blockIdx.x * 256 + threadIdx.x) * 8;
  float4 v0 = *(const float4*)(src + i);
  float4 v1 = *(const float4*)(src + i + 4);
  v8bf pk;
  pk[0] = f2bf(v0.x); pk[1] = f2bf(v0.y); pk[2] = f2bf(v0.z); pk[3] = f2bf(v0.w);
  pk[4] = f2bf(v1.x); pk[5] = f2bf(v1.y); pk[6] = f2bf(v1.z); pk[7] = f2bf(v1.w);
  *(v8bf*)(dst + i) = pk;
}

// ---------------------------------------------------------------------------
// Kernel 1: fused QKV projection on bf16 sources.  Y = X @ W^T + bias.
// Tile: 128(M) x 128(N) per 8-wave block, K-step 64.  TDM double-buffering:
// waves 0/1 issue the NEXT iteration's X/W tile DMAs before waiting on the
// current ones (TENSORcnt <= 1), overlapping DMA with WMMA compute.
// ---------------------------------------------------------------------------
__global__ __launch_bounds__(256) void qkv_proj_kernel(
    const bf16_t* __restrict__ Xb, const bf16_t* __restrict__ Wb,
    const float* __restrict__ bq, const float* __restrict__ bk,
    const float* __restrict__ bv,
    bf16_t* __restrict__ Qh, bf16_t* __restrict__ Kh, bf16_t* __restrict__ Vt)
{
  // 64 KB staging: 2 ping-pong buffers x (X tile, W tile), each 128x64 bf16.
  // First 16 KB reused as a 128x64 transpose buffer in the Q/K epilogue.
  __shared__ __align__(16) bf16_t ls[4 * 128 * 64];

  const int z = blockIdx.z;
  const bf16_t* W   = Wb + (size_t)z * D_MODEL * D_MODEL;
  const float* bias = (z == 0) ? bq : ((z == 1) ? bk : bv);

  const int m0   = blockIdx.x * 128;
  const int n0   = blockIdx.y * 128;
  const int tid  = threadIdx.x;
  const int lane = tid & 31;
  // Scalarized wave id: turns the DMA-issue guards into s_cbranch.
  const int wv   = __builtin_amdgcn_readfirstlane(tid >> 5);
  const int waveM = wv >> 1;  // 0..3 -> M offset waveM*32
  const int waveN = wv & 1;   // 0..1 -> N offset waveN*64

  v8f acc[2][4];
#pragma unroll
  for (int i = 0; i < 2; ++i)
#pragma unroll
    for (int j = 0; j < 4; ++j) acc[i][j] = zero8();

  const int NK = D_MODEL / 64;                    // 16 K-steps
  const bf16_t* gx = Xb + (size_t)m0 * D_MODEL;   // X tile row base
  const bf16_t* gw = W + (size_t)n0 * D_MODEL;    // W tile row base

#ifdef USE_TDM
  const unsigned lds0 = (unsigned)(size_t)(void*)ls;  // flat LDS addr[31:0]
  // Prologue: kick off tile 0 DMAs into buffer 0.
  if (wv == 0) tdm_load_tile(gx, lds0, BATCH * SEQ);
  else if (wv == 1) tdm_load_tile(gw, lds0 + 16384, D_MODEL);
#endif

  for (int it = 0; it < NK; ++it) {
    bf16_t* lx = ls + (it & 1) * (2 * 128 * 64);
    bf16_t* lw = lx + 128 * 64;
#ifdef USE_TDM
    // Issue next tile into the other buffer, then wait for the current one
    // (per-wave TDM ops complete in order -> TENSORcnt <= 1 retires tile it).
    if (it + 1 < NK) {
      const unsigned nb = lds0 + ((it + 1) & 1) * 32768;
      if (wv == 0) {
        tdm_load_tile(gx + (it + 1) * 64, nb, BATCH * SEQ);
        __builtin_amdgcn_s_wait_tensorcnt(1);
      } else if (wv == 1) {
        tdm_load_tile(gw + (it + 1) * 64, nb + 16384, D_MODEL);
        __builtin_amdgcn_s_wait_tensorcnt(1);
      }
    } else if (wv < 2) {
      __builtin_amdgcn_s_wait_tensorcnt(0);
    }
#else
    {
      const int row  = tid >> 1;          // 0..127
      const int colb = (tid & 1) * 32;    // 0 or 32
      const bf16_t* sx = gx + (size_t)row * D_MODEL + it * 64 + colb;
      const bf16_t* sw = gw + (size_t)row * D_MODEL + it * 64 + colb;
#pragma unroll
      for (int c = 0; c < 32; c += 8) {
        *(v8bf*)(lx + row * 64 + colb + c) = *(const v8bf*)(sx + c);
        *(v8bf*)(lw + row * 64 + colb + c) = *(const v8bf*)(sw + c);
      }
    }
#endif
    __syncthreads();

#pragma unroll
    for (int kk = 0; kk < 2; ++kk) {      // two 32-wide K slices
      v16bf a[2], bw[4];
#pragma unroll
      for (int i = 0; i < 2; ++i)
        a[i] = load_frag(lx + (waveM * 32 + i * 16) * 64 + kk * 32, 64, lane);
#pragma unroll
      for (int j = 0; j < 4; ++j)
        bw[j] = load_frag(lw + (waveN * 64 + j * 16) * 64 + kk * 32, 64, lane);
#pragma unroll
      for (int i = 0; i < 2; ++i)
#pragma unroll
        for (int j = 0; j < 4; ++j)
          acc[i][j] = wmma_bf16(a[i], bw[j], acc[i][j]);
    }
    __syncthreads();
  }

  const int nlane = lane & 15;
  const int mh    = (lane >> 4) * 8;

  if (z == 2) {
    // V epilogue: Vt[b,h,hd,s].  For fixed (i,j,lane) the 8 accumulator rows
    // are consecutive in s -> one packed 16-byte store each.
#pragma unroll
    for (int i = 0; i < 2; ++i) {
#pragma unroll
      for (int j = 0; j < 4; ++j) {
        const int N  = n0 + waveN * 64 + j * 16 + nlane;
        const float bval = bias[N];
        const int h  = N >> 6;
        const int hd = N & 63;
        const int mb = m0 + waveM * 32 + i * 16 + mh;  // tile fits one batch
        const int b  = mb >> 11;
        const int s0 = mb & 2047;
        v8bf pk;
#pragma unroll
        for (int r = 0; r < 8; ++r) pk[r] = f2bf(acc[i][j][r] + bval);
        *(v8bf*)(Vt + (size_t)((b * N_HEAD + h) * HEAD_DIM + hd) * SEQ + s0) = pk;
      }
    }
  } else {
    // Q/K epilogue: transpose 128x64 half-tiles (= one head's hd range)
    // through LDS, then coalesced 64-wide row stores.
    bf16_t* dst = (z == 0) ? Qh : Kh;
    const float scale = (z == 0) ? 0.125f : 1.0f;  // fold 1/sqrt(hd) into Q
#pragma unroll
    for (int pass = 0; pass < 2; ++pass) {
      if (waveN == pass) {
#pragma unroll
        for (int i = 0; i < 2; ++i) {
#pragma unroll
          for (int j = 0; j < 4; ++j) {
            const int N  = n0 + pass * 64 + j * 16 + nlane;
            const float bval = bias[N];
            const int row0 = waveM * 32 + i * 16 + mh;
#pragma unroll
            for (int r = 0; r < 8; ++r)
              ls[(row0 + r) * 64 + j * 16 + nlane] =
                  f2bf((acc[i][j][r] + bval) * scale);
          }
        }
      }
      __syncthreads();
      {
        const int h    = (n0 >> 6) + pass;
        const int row  = tid >> 1;           // 0..127
        const int colb = (tid & 1) * 32;     // 0 or 32
        const int m = m0 + row;
        const int b = m >> 11;
        const int s = m & 2047;
        bf16_t* o = dst + (size_t)((b * N_HEAD + h) * SEQ + s) * HEAD_DIM + colb;
        const bf16_t* src = ls + row * 64 + colb;
#pragma unroll
        for (int c = 0; c < 32; c += 8)
          *(v8bf*)(o + c) = *(const v8bf*)(src + c);
      }
      __syncthreads();
    }
  }
}

// ---------------------------------------------------------------------------
// Kernel 2: flash attention.  One wave per (b, h, 16-query tile), 64-key
// blocks.  Online softmax; P round-trips through a private per-wave LDS tile
// (per-wave DS ordering makes barriers unnecessary).  Context (fp32) written
// directly into d_out at [b, s, h*64 + hd].
// ---------------------------------------------------------------------------
__global__ __launch_bounds__(128) void attn_kernel(
    const bf16_t* __restrict__ Qh, const bf16_t* __restrict__ Kh,
    const bf16_t* __restrict__ Vt, float* __restrict__ Ctx)
{
  __shared__ __align__(16) bf16_t P[4][16 * 64];  // 2 KB per wave

  const int lane = threadIdx.x & 31;
  // Scalarized: all per-wave bases become SGPRs.
  const int wave = __builtin_amdgcn_readfirstlane(threadIdx.x >> 5);
  const int gw   = blockIdx.x * 4 + wave;   // global q-tile id, 0..8191
  const int b    = gw >> 11;                // / (N_HEAD * SEQ/16)
  const int h    = (gw >> 7) & 15;
  const int qt   = gw & 127;

  const bf16_t* Qbase = Qh + (size_t)((b * N_HEAD + h) * SEQ + qt * 16) * HEAD_DIM;
  const bf16_t* Kbase = Kh + (size_t)((b * N_HEAD + h) * SEQ) * HEAD_DIM;
  const bf16_t* Vbase = Vt + (size_t)((b * N_HEAD + h) * HEAD_DIM) * SEQ;

  // Q fragments (reused for every key block)
  v16bf q0 = load_frag(Qbase, HEAD_DIM, lane);        // hd 0..31
  v16bf q1 = load_frag(Qbase + 32, HEAD_DIM, lane);   // hd 32..63

  v8f c[4];
#pragma unroll
  for (int j = 0; j < 4; ++j) c[j] = zero8();
  float mrow[8], lrow[8];
#pragma unroll
  for (int r = 0; r < 8; ++r) { mrow[r] = -3.0e38f; lrow[r] = 0.0f; }

  bf16_t* Pme   = &P[wave][0];
  const int nl  = lane & 15;
  const int mh8 = (lane >> 4) << 3;

  for (int kb = 0; kb < SEQ; kb += 64) {
    const bf16_t* kblk = Kbase + (size_t)kb * HEAD_DIM;

    // Scores for 4 key sub-tiles: S = Q K^T (k-dim 64 in two WMMA steps).
    v8f sc[4];
#pragma unroll
    for (int t = 0; t < 4; ++t) {
      v16bf klo = load_frag(kblk + (size_t)(t * 16) * HEAD_DIM, HEAD_DIM, lane);
      v16bf khi = load_frag(kblk + (size_t)(t * 16) * HEAD_DIM + 32, HEAD_DIM, lane);
      sc[t] = wmma_bf16(q0, klo, zero8());
      sc[t] = wmma_bf16(q1, khi, sc[t]);
    }

    // Online softmax update (row M = r + 8*(lane>>4); cols striped over the
    // 16 lanes of each half-wave, so xor-masks 1..8 reduce a full row).
#pragma unroll
    for (int r = 0; r < 8; ++r) {
      float bm = fmaxf(fmaxf(sc[0][r], sc[1][r]), fmaxf(sc[2][r], sc[3][r]));
#pragma unroll
      for (int mask = 1; mask < 16; mask <<= 1)
        bm = fmaxf(bm, __shfl_xor(bm, mask, 32));
      const float mnew = fmaxf(mrow[r], bm);
      const float corr = __expf(mrow[r] - mnew);
      mrow[r] = mnew;
      float p[4];
#pragma unroll
      for (int t = 0; t < 4; ++t) p[t] = __expf(sc[t][r] - mnew);
      float bs = (p[0] + p[1]) + (p[2] + p[3]);
#pragma unroll
      for (int mask = 1; mask < 16; mask <<= 1)
        bs += __shfl_xor(bs, mask, 32);
      lrow[r] = lrow[r] * corr + bs;
#pragma unroll
      for (int j = 0; j < 4; ++j) c[j][r] *= corr;

      const int M = r + mh8;
#pragma unroll
      for (int t = 0; t < 4; ++t)
        Pme[M * 64 + t * 16 + nl] = f2bf(p[t]);
    }

    // ctx += P @ V.  Per-wave LDS region + in-order DS guarantees ordering.
    v16bf pa0 = load_frag(Pme, 64, lane);        // keys kb+0 .. kb+31
    v16bf pa1 = load_frag(Pme + 32, 64, lane);   // keys kb+32 .. kb+63
#pragma unroll
    for (int j = 0; j < 4; ++j) {
      v16bf bv0 = load_frag(Vbase + (size_t)(j * 16) * SEQ + kb, SEQ, lane);
      v16bf bv1 = load_frag(Vbase + (size_t)(j * 16) * SEQ + kb + 32, SEQ, lane);
      c[j] = wmma_bf16(pa0, bv0, c[j]);
      c[j] = wmma_bf16(pa1, bv1, c[j]);
    }
  }

  // Normalize and store fp32 context.
#pragma unroll
  for (int r = 0; r < 8; ++r) {
    const float inv = 1.0f / lrow[r];
    const int s = qt * 16 + r + mh8;
#pragma unroll
    for (int j = 0; j < 4; ++j) {
      const int d = h * HEAD_DIM + j * 16 + nl;
      Ctx[(size_t)(b * SEQ + s) * D_MODEL + d] = c[j][r] * inv;
    }
  }
}

// ---------------------------------------------------------------------------
// Kernel 3: residual add + LayerNorm, in place over d_out.
// One 256-thread block per token; 4 elements per thread.
// ---------------------------------------------------------------------------
__global__ __launch_bounds__(256) void ln_kernel(
    float* __restrict__ Y, const float* __restrict__ X,
    const float* __restrict__ gamma, const float* __restrict__ beta)
{
  __shared__ float ssum[8], ssq[8];
  const int tid  = threadIdx.x;
  const size_t base = (size_t)blockIdx.x * D_MODEL;

  float4 y = *(const float4*)(Y + base + tid * 4);
  float4 x = *(const float4*)(X + base + tid * 4);
  y.x += x.x; y.y += x.y; y.z += x.z; y.w += x.w;

  float s = y.x + y.y + y.z + y.w;
  float q = y.x * y.x + y.y * y.y + y.z * y.z + y.w * y.w;
#pragma unroll
  for (int mask = 1; mask < 32; mask <<= 1) {
    s += __shfl_xor(s, mask, 32);
    q += __shfl_xor(q, mask, 32);
  }
  const int lane = tid & 31, wave = tid >> 5;
  if (lane == 0) { ssum[wave] = s; ssq[wave] = q; }
  __syncthreads();
  float S = 0.0f, Q = 0.0f;
#pragma unroll
  for (int i = 0; i < 8; ++i) { S += ssum[i]; Q += ssq[i]; }

  const float mu   = S * (1.0f / D_MODEL);
  const float var  = Q * (1.0f / D_MODEL) - mu * mu;
  const float rstd = rsqrtf(var + 1e-5f);

  float4 g  = *(const float4*)(gamma + tid * 4);
  float4 be = *(const float4*)(beta + tid * 4);
  float4 o;
  o.x = (y.x - mu) * rstd * g.x + be.x;
  o.y = (y.y - mu) * rstd * g.y + be.y;
  o.z = (y.z - mu) * rstd * g.z + be.z;
  o.w = (y.w - mu) * rstd * g.w + be.w;
  *(float4*)(Y + base + tid * 4) = o;
}

// ---------------------------------------------------------------------------
extern "C" void kernel_launch(void* const* d_in, const int* in_sizes, int n_in,
                              void* d_out, int out_size, void* d_ws, size_t ws_size,
                              hipStream_t stream) {
  (void)in_sizes; (void)n_in; (void)out_size; (void)ws_size;

  const float* x     = (const float*)d_in[0];
  const float* Wq    = (const float*)d_in[1];
  const float* bq    = (const float*)d_in[2];
  const float* Wk    = (const float*)d_in[3];
  const float* bk    = (const float*)d_in[4];
  const float* Wv    = (const float*)d_in[5];
  const float* bv    = (const float*)d_in[6];
  const float* gamma = (const float*)d_in[7];
  const float* beta  = (const float*)d_in[8];
  float* out = (float*)d_out;

  // Workspace layout (bytes):
  //   Qh  @ 0     : 16 MB  bf16 [b,h,s,hd]
  //   Kh  @ 16 MB : 16 MB  bf16 [b,h,s,hd]
  //   Vt  @ 32 MB : 16 MB  bf16 [b,h,hd,s]
  //   Xb  @ 48 MB : 16 MB  bf16 [b*s, d]
  //   Wb  @ 64 MB :  6 MB  bf16 [3, d, d]
  char* ws = (char*)d_ws;
  const size_t elems = (size_t)BATCH * SEQ * D_MODEL;
  const size_t welems = (size_t)D_MODEL * D_MODEL;
  bf16_t* Qh = (bf16_t*)(ws);
  bf16_t* Kh = (bf16_t*)(ws + elems * 2);
  bf16_t* Vt = (bf16_t*)(ws + 2 * elems * 2);
  bf16_t* Xb = (bf16_t*)(ws + 3 * elems * 2);
  bf16_t* Wb = (bf16_t*)(ws + 4 * elems * 2);

  // Pass 0: one-time fp32 -> bf16 conversion of X and the three weights.
  cvt_kernel<<<(unsigned)(elems / 2048), 256, 0, stream>>>(x, Xb);
  cvt_kernel<<<(unsigned)(welems / 2048), 256, 0, stream>>>(Wq, Wb);
  cvt_kernel<<<(unsigned)(welems / 2048), 256, 0, stream>>>(Wk, Wb + welems);
  cvt_kernel<<<(unsigned)(welems / 2048), 256, 0, stream>>>(Wv, Wb + 2 * welems);

  dim3 g1(BATCH * SEQ / 128, D_MODEL / 128, 3);
  qkv_proj_kernel<<<g1, 256, 0, stream>>>(Xb, Wb, bq, bk, bv, Qh, Kh, Vt);

  dim3 g2(BATCH * N_HEAD * (SEQ / 16) / 4);
  attn_kernel<<<g2, 128, 0, stream>>>(Qh, Kh, Vt, out);

  ln_kernel<<<BATCH * SEQ, 256, 0, stream>>>(out, x, gamma, beta);
}
